// LangevinSampler_53317724013243
// MI455X (gfx1250) — matrix operand order repaired and checked by reading, not compiled, not run on hardware
//
#include <hip/hip_runtime.h>
#include <hip/hip_bf16.h>
#include <stdint.h>

// ---------------- problem constants (from reference setup) ----------------
#define B_DIM 16
#define S_DIM 64
#define P_DIM 16
#define L_DIM 48            // S - P
#define V_DIM 50257
#define K_TOP 50
#define E_DIM 768
#define M_DIM (B_DIM * L_DIM)   // 768 GEMM rows
#define EPSV  1e-10f
#define WVAL  5.0f

typedef __bf16 bf16_t;
typedef __attribute__((ext_vector_type(16))) __bf16 v16bf;
typedef __attribute__((ext_vector_type(8)))  __bf16 v8bf;
typedef __attribute__((ext_vector_type(8)))  float  v8f;
typedef __attribute__((ext_vector_type(4)))  float  v4f;

// monotone float->uint mapping: order-preserving for descending top-k
__device__ __forceinline__ unsigned mono(float f) {
  unsigned u = __float_as_uint(f);
  return (u & 0x80000000u) ? ~u : (u | 0x80000000u);
}

// round-to-nearest-even float -> bf16 without relying on cast lowering
__device__ __forceinline__ bf16_t f2bf(float f) {
  unsigned u = __float_as_uint(f);
  unsigned r = (u + 0x7FFFu + ((u >> 16) & 1u)) >> 16;
  unsigned short us = (unsigned short)r;
  return __builtin_bit_cast(bf16_t, us);
}

// =====================================================================
// Kernel 1: exact top-50 via 4-pass radix select + gumbel argmax sample
// one block (256 threads) per (b,l) row
// =====================================================================
__global__ __launch_bounds__(256) void topk_sample_kernel(
    const float* __restrict__ gx,       // (B,L,V)
    const float* __restrict__ logits,   // (B,S,V)
    const int*   __restrict__ cur_ids,  // (B,S)
    const float* __restrict__ gumbel,   // (B,L,K)
    int* __restrict__ out_ids,          // ws: (M)
    int* __restrict__ out_ids2)         // d_out tail: (M)
{
  __shared__ unsigned hist[256];
  __shared__ unsigned sh_prefix;
  __shared__ int      sh_krem;
  __shared__ int      sh_cnt;
  __shared__ int      sh_M;
  __shared__ unsigned cu[64];
  __shared__ int      ci[64];
  __shared__ float    sc[64];

  const int row = blockIdx.x;               // 0..M-1
  const int b   = row / L_DIM;
  const int l   = row % L_DIM;
  const int tid = threadIdx.x;
  const float* lrow = logits + (size_t)(b * S_DIM + P_DIM + l) * V_DIM;

  if (tid == 0) { sh_prefix = 0u; sh_krem = K_TOP; sh_cnt = 0; }
  __syncthreads();

  // 4 radix passes, MSB first, 8-bit digits -> exact Kth-largest threshold
  for (int shift = 24; shift >= 0; shift -= 8) {
    hist[tid] = 0u;
    __syncthreads();
    const unsigned prefix = sh_prefix;
    const unsigned pmask  = (shift == 24) ? 0u : (0xFFFFFFFFu << (shift + 8));
    for (int v = tid; v < V_DIM; v += 256) {
      unsigned u = mono(lrow[v]);
      if ((u & pmask) == prefix)
        atomicAdd(&hist[(u >> shift) & 255u], 1u);
    }
    __syncthreads();
    if (tid == 0) {
      int krem = sh_krem;
      unsigned d = 255u;
      for (;;) {
        int c = (int)hist[d];
        if (c >= krem) break;
        krem -= c;
        if (d == 0u) break;
        --d;
      }
      sh_krem   = krem;
      sh_prefix = prefix | (d << shift);
    }
    __syncthreads();
  }

  // collect all candidates >= threshold (== exactly K for distinct floats)
  const unsigned thresh = sh_prefix;
  for (int v = tid; v < V_DIM; v += 256) {
    unsigned u = mono(lrow[v]);
    if (u >= thresh) {
      int p = atomicAdd(&sh_cnt, 1);
      if (p < 64) { cu[p] = u; ci[p] = v; }
    }
  }
  __syncthreads();

  // sort candidates: descending value, ascending index (jax top_k order)
  if (tid == 0) {
    int M = sh_cnt; if (M > 64) M = 64;
    for (int i = 1; i < M; ++i) {
      unsigned ku = cu[i]; int ki = ci[i];
      int j = i - 1;
      while (j >= 0 && (cu[j] < ku || (cu[j] == ku && ci[j] > ki))) {
        cu[j + 1] = cu[j]; ci[j + 1] = ci[j]; --j;
      }
      cu[j + 1] = ku; ci[j + 1] = ki;
    }
    sh_M = (M < K_TOP) ? M : K_TOP;
  }
  __syncthreads();

  // gumbel-perturbed scores on the top-K (gx only gathered, never streamed)
  const int M = sh_M;
  const int curid = cur_ids[b * S_DIM + P_DIM + l];
  if (tid < M) {
    int id = ci[tid];
    float dl = -gx[(size_t)row * V_DIM + id] * (id == curid ? EPSV : 1.0f);
    sc[tid] = dl + gumbel[(size_t)row * K_TOP + tid];
  }
  __syncthreads();
  if (tid == 0) {
    int best = 0; float bv = sc[0];
    for (int k = 1; k < M; ++k) { if (sc[k] > bv) { bv = sc[k]; best = k; } }
    int aid = ci[best];
    out_ids[row]  = aid;
    out_ids2[row] = aid;   // second output of the tuple (raw int32 bits)
  }
}

// =====================================================================
// Kernel 2: t1[v] = sum_e W[v,e]^2   (one wave of 32 per row)
// =====================================================================
__global__ __launch_bounds__(256) void t1_kernel(
    const float* __restrict__ W, float* __restrict__ t1)
{
  const int wid  = blockIdx.x * 8 + (threadIdx.x >> 5);
  const int lane = threadIdx.x & 31;
  if (wid >= V_DIM) return;                     // wave-uniform exit
  const float* r = W + (size_t)wid * E_DIM;
  float s = 0.f;
  for (int e = lane; e < E_DIM; e += 32) { float f = r[e]; s += f * f; }
  for (int m = 16; m > 0; m >>= 1) s += __shfl_xor(s, m, 32);
  if (lane == 0) t1[wid] = s;
}

// =====================================================================
// Kernel 3: gather cur_embeds -> bf16 A matrix + t3 row norms
// one block (256 threads) per GEMM row
// =====================================================================
__global__ __launch_bounds__(256) void gather_kernel(
    const float* __restrict__ W, const int* __restrict__ ids,
    bf16_t* __restrict__ A, float* __restrict__ t3)
{
  __shared__ float red[256];
  const int row = blockIdx.x;
  const int tid = threadIdx.x;
  const int id  = ids[row];
  const float* r = W + (size_t)id * E_DIM;
  float s = 0.f;
  for (int e = tid; e < E_DIM; e += 256) {
    float f = r[e];
    A[(size_t)row * E_DIM + e] = f2bf(f);
    s += f * f;
  }
  red[tid] = s; __syncthreads();
  for (int st = 128; st > 0; st >>= 1) {
    if (tid < st) red[tid] += red[tid + st];
    __syncthreads();
  }
  if (tid == 0) t3[row] = red[0];
}

// =====================================================================
// Kernel 4: bias = 10*(A x W^T) - 5*(t1[n] + t3[m]), bf16 WMMA GEMM
// block tile 128Mx128N, K-step 32; 8 waves each compute 64Mx32N.
// Double-buffered LDS; A tile staged memory->LDS with async-LDS loads
// (ASYNCcnt), B tile staged via VGPRs with inline fp32->bf16 convert.
// Last k-step peeled so the steady-state loop body is branch-free.
// =====================================================================
#define BM 128
#define BN 128
#define BK 32
#define LDSS 40   // LDS row stride in bf16 elems (80B: 16B-aligned, conflict-free)

__global__ __launch_bounds__(256) void gemm_bias_kernel(
    const bf16_t* __restrict__ A,    // (M,E) bf16 row-major
    const float*  __restrict__ W,    // (V,E) fp32 row-major  == B^T
    const float*  __restrict__ t1,
    const float*  __restrict__ t3,
    float* __restrict__ out)         // (M,V)
{
  __shared__ __align__(16) bf16_t sA[2][BM * LDSS];
  __shared__ __align__(16) bf16_t sB[2][BN * LDSS];

  const int tid  = threadIdx.x;
  const int n0   = blockIdx.x * BN;
  const int m0   = blockIdx.y * BM;
  const int wave = tid >> 5;           // 0..7
  const int lane = tid & 31;
  const int wm   = wave >> 2;          // 0..1 : 64-row slab
  const int wn   = wave & 3;           // 0..3 : 32-col slab

  v8f acc[4][2];
  for (int i = 0; i < 4; ++i)
    for (int j = 0; j < 2; ++j)
      for (int h = 0; h < 8; ++h) acc[i][j][h] = 0.f;

  const int srow  = tid >> 1;   // 0..127: staged row
  const int shalf = tid & 1;    // which 16-element half of the 32-wide k slab

  // loop-invariant staging pointers; clamp OOB B rows (their columns are
  // never stored, and WMMA columns are independent -> no zero-fill needed)
  const bf16_t* gA = A + (size_t)(m0 + srow) * E_DIM + shalf * 16;
  const int nclamp = ((n0 + srow) < V_DIM) ? (n0 + srow) : (V_DIM - 1);
  const float* gB  = W + (size_t)nclamp * E_DIM + shalf * 16;

  // A tile: pure bf16 copy -> async global->LDS (no VGPR round trip)
  auto stageA = [&](int kt, int buf) {
    unsigned l0 = (unsigned)(size_t)&sA[buf][srow * LDSS + shalf * 16];
    const void* g = (const void*)(gA + kt);
    asm volatile("global_load_async_to_lds_b128 %0, %1, off"
                 :: "v"(l0), "v"(g) : "memory");
    asm volatile("global_load_async_to_lds_b128 %0, %1, off offset:16"
                 :: "v"(l0), "v"(g) : "memory");
  };
  // B tile: fp32 -> bf16 inline conversion through VGPRs
  auto stageB = [&](int kt, int buf, bool pf) {
    const v4f* src = reinterpret_cast<const v4f*>(gB + kt);
    v4f q0 = src[0], q1 = src[1], q2 = src[2], q3 = src[3];
    if (pf) __builtin_prefetch(gB + kt + BK, 0, 1);
    v8bf o0, o1;
    for (int h = 0; h < 4; ++h) { o0[h] = f2bf(q0[h]); o0[h + 4] = f2bf(q1[h]); }
    for (int h = 0; h < 4; ++h) { o1[h] = f2bf(q2[h]); o1[h + 4] = f2bf(q3[h]); }
    v8bf* dst = reinterpret_cast<v8bf*>(&sB[buf][srow * LDSS + shalf * 16]);
    dst[0] = o0; dst[1] = o1;
  };

  const int hiA  = (lane >> 4) * 8;    // A frag: lanes 16-31 hold K+8 / K+24
  const int kSel = (lane >> 4) * 8;    // B frag: lanes 16-31 hold K+8 / K+24
  const int mrow = lane & 15;

  // fragment-load + 8x WMMA on one LDS buffer
  auto compute = [&](int buf) {
    v16bf bfr[2], afr[4];
    for (int jn = 0; jn < 2; ++jn) {
      const int n = wn * 32 + jn * 16 + mrow;
      const v8bf* p = reinterpret_cast<const v8bf*>(&sB[buf][n * LDSS + kSel]);
      v8bf lo = p[0], hi = p[2];
      for (int h = 0; h < 8; ++h) { bfr[jn][h] = lo[h]; bfr[jn][h + 8] = hi[h]; }
    }
    for (int im = 0; im < 4; ++im) {
      const int m = wm * 64 + im * 16 + mrow;
      const v8bf* p0 = reinterpret_cast<const v8bf*>(&sA[buf][m * LDSS + hiA]);
      const v8bf* p1 = reinterpret_cast<const v8bf*>(&sA[buf][m * LDSS + 16 + hiA]);
      v8bf lo = p0[0], hi = p1[0];
      for (int h = 0; h < 8; ++h) { afr[im][h] = lo[h]; afr[im][h + 8] = hi[h]; }
    }
    for (int im = 0; im < 4; ++im)
      for (int jn = 0; jn < 2; ++jn)
        acc[im][jn] = __builtin_amdgcn_wmma_f32_16x16x32_bf16(
            false, afr[im], false, bfr[jn], (short)0, acc[im][jn], false, false);
  };

  // prologue: fill buffer 0
  stageA(0, 0);
  stageB(0, 0, true);
  asm volatile("s_wait_asynccnt 0x0" ::: "memory");
  __syncthreads();

  // steady state: unconditionally stage next tile, compute current
  int buf = 0;
  for (int kt = 0; kt < E_DIM - BK; kt += BK, buf ^= 1) {
    stageA(kt + BK, buf ^ 1);
    stageB(kt + BK, buf ^ 1, kt + 2 * BK < E_DIM);
    compute(buf);
    // my async A-writes for the next buffer must land before the barrier
    asm volatile("s_wait_asynccnt 0x0" ::: "memory");
    __syncthreads();
  }
  // peeled final k-step: nothing left to stage, no barrier needed after
  compute(buf);

  // ---- fused epilogue: bias = 10*acc - 5*(t1[n] + t3[m]) ----
  for (int im = 0; im < 4; ++im) {
    for (int jn = 0; jn < 2; ++jn) {
      const int n = n0 + wn * 32 + jn * 16 + (lane & 15);
      if (n >= V_DIM) continue;
      const float tn = t1[n];
      const int mbase = m0 + wm * 64 + im * 16 + (lane >> 4) * 8;
      for (int j = 0; j < 8; ++j) {
        const int m = mbase + j;
        out[(size_t)m * V_DIM + n] = 10.f * acc[im][jn][j] - WVAL * (tn + t3[m]);
      }
    }
  }
}

// =====================================================================
extern "C" void kernel_launch(void* const* d_in, const int* in_sizes, int n_in,
                              void* d_out, int out_size, void* d_ws, size_t ws_size,
                              hipStream_t stream)
{
  const float* gx      = (const float*)d_in[0];
  const float* logits  = (const float*)d_in[1];
  const int*   cur_ids = (const int*)  d_in[2];
  const float* W       = (const float*)d_in[3];
  const float* gumbel  = (const float*)d_in[4];
  // d_in[5] = prompt_length scalar (fixed 16, baked into P_DIM)
  (void)in_sizes; (void)n_in; (void)out_size; (void)ws_size;

  char*   ws   = (char*)d_ws;
  int*    ids  = (int*)  (ws + 0);            // 768 * 4  = 3 KB
  float*  t3   = (float*)(ws + 4096);         // 768 * 4  = 3 KB
  float*  t1   = (float*)(ws + 8192);         // 50257 * 4 ~ 197 KB
  bf16_t* A    = (bf16_t*)(ws + 212992);      // 768*768*2 ~ 1.13 MB

  float* bias   = (float*)d_out;                                  // (M,V)
  int*   outIds = (int*)((float*)d_out + (size_t)M_DIM * V_DIM);  // (M,) int32 bits

  topk_sample_kernel<<<dim3(M_DIM), dim3(256), 0, stream>>>(
      gx, logits, cur_ids, gumbel, ids, outIds);
  t1_kernel<<<dim3((V_DIM + 7) / 8), dim3(256), 0, stream>>>(W, t1);
  gather_kernel<<<dim3(M_DIM), dim3(256), 0, stream>>>(W, ids, A, t3);
  gemm_bias_kernel<<<dim3((V_DIM + BN - 1) / BN, M_DIM / BM), dim3(256), 0, stream>>>(
      A, W, t1, t3, bias);
}
